// MMContextEncoderBiDirCrossAttn_10411000726057
// MI455X (gfx1250) — compile-verified, matrix operand order checked by reading, not run on hardware
//
#include <hip/hip_runtime.h>
#include <math.h>

// ---------------- problem constants (from reference) ----------------
#define NS   4
#define BT   10       // B*T = 2*5
#define C    384
#define CTX  128
#define DH   64
#define SCALE 0.125f  // DH^-0.5

typedef __attribute__((ext_vector_type(16))) __bf16 v16bf;
typedef __attribute__((ext_vector_type(8)))  float  v8f;
typedef __attribute__((ext_vector_type(4)))  float  v4f;

// float -> bf16 bits, round-to-nearest-even
__device__ __forceinline__ unsigned short f2bf(float f) {
    unsigned int u = __float_as_uint(f);
    u = (u + 0x7FFFu + ((u >> 16) & 1u)) >> 16;
    return (unsigned short)u;
}

// ---------------------------------------------------------------
// Kernel P: repack qk_w / v_w (NS, DH, C) f32 -> bf16 in the WMMA
// B-matrix (K=32 x N=16) lane layout, per (stage, matrix, kchunk, ntile).
// flat id = ((((s*2+mat)*12 + kc)*4 + nt)*32 + lane)*16 + e
// ---------------------------------------------------------------
__global__ __launch_bounds__(256)
void repack_kernel(const float* __restrict__ qk_w,
                   const float* __restrict__ v_w,
                   unsigned short* __restrict__ brep)
{
    int id   = blockIdx.x * 256 + threadIdx.x;   // exactly NS*2*12*4*32*16 = 196608
    int e    = id & 15;
    int lane = (id >> 4) & 31;
    int nt   = (id >> 9) & 3;
    int rest = id >> 11;
    int kc   = rest % 12;
    int ms   = rest / 12;
    int mat  = ms & 1;
    int s    = ms >> 1;
    int K    = ((lane >> 4) << 3) + e + (e < 8 ? 0 : 8);   // K within 32-chunk
    int c    = (kc << 5) + K;
    int d    = (nt << 4) + (lane & 15);
    const float* W = (mat == 0) ? qk_w : v_w;
    brep[id] = f2bf(W[(size_t)(s * DH + d) * C + c]);
}

// ---------------------------------------------------------------
// Kernel A: per (stage, bt) small dense math.
//   a      = audio @ Wp.T + bp                       (C)
//   cqk    = (a + pe[bt%5]) @ Wcqk.T                 (DH)  -> ws
//   cv     = a @ Wcv.T                               (DH)
//   outrow = cv @ Wo.T + bo                          (C)   -> ws
// ---------------------------------------------------------------
__global__ __launch_bounds__(384)
void prep_kernel(const float* __restrict__ au0, const float* __restrict__ au1,
                 const float* __restrict__ au2, const float* __restrict__ au3,
                 const float* __restrict__ Wp,  const float* __restrict__ bp,
                 const float* __restrict__ pe,
                 const float* __restrict__ Wcqk, const float* __restrict__ Wcv,
                 const float* __restrict__ Wo,   const float* __restrict__ bo,
                 float* __restrict__ cqk_ws, float* __restrict__ outrow_ws)
{
    const int s  = blockIdx.x;
    const int bt = blockIdx.y;
    const int t  = threadIdx.x;
    __shared__ float a_sh[C];
    __shared__ float cv_sh[DH];

    const float* audio = (s == 0 ? au0 : s == 1 ? au1 : s == 2 ? au2 : au3) + bt * CTX;
    float acc = bp[s * C + t];
    const float* wrow = Wp + (size_t)(s * C + t) * CTX;
    for (int k = 0; k < CTX; ++k) acc += audio[k] * wrow[k];
    a_sh[t] = acc;
    __syncthreads();

    if (t < DH) {
        const float* per = pe + (s * 5 + (bt % 5)) * C;
        const float* wq  = Wcqk + (size_t)(s * DH + t) * C;
        const float* wv  = Wcv  + (size_t)(s * DH + t) * C;
        float q = 0.f, vv = 0.f;
        for (int c = 0; c < C; ++c) {
            float av = a_sh[c];
            q  += (av + per[c]) * wq[c];
            vv += av * wv[c];
        }
        cqk_ws[(s * BT + bt) * DH + t] = q;
        cv_sh[t] = vv;
    }
    __syncthreads();

    float o = bo[s * C + t];
    const float* wo = Wo + (size_t)(s * C + t) * DH;
    for (int d = 0; d < DH; ++d) o += cv_sh[d] * wo[d];
    outrow_ws[(s * BT + bt) * C + t] = o;
}

// ---------------------------------------------------------------
// Kernel B: the heavy streaming GEMM per stage (templated on HW so the
// channel stride is a compile-time constant -> A-tile gathers become
// immediate-offset load clauses with no 64-bit address chains).
// One wave: 16 positions x DH=64 outputs for both qk and v:
//   qk = (x+pos) @ Wqk.T ;  v = x @ Wv.T   (K = C = 384, 12 chunks of 32)
// bf16 WMMA, f32 accumulate; sim = SCALE * qk . cqk via shfl reduce.
// ---------------------------------------------------------------
template <int HW>
__global__ __launch_bounds__(128)
void stage_qkv_kernel(const float* __restrict__ x, const float* __restrict__ ps,
                      const unsigned short* __restrict__ bq,   // stage's repacked Wqk
                      const unsigned short* __restrict__ bv,   // stage's repacked Wv
                      const float* __restrict__ cqk,           // stage base: [bt][DH]
                      float* __restrict__ sim,                 // [bt][hw]
                      float* __restrict__ vws)                 // [bt][hw][DH]
{
    const int wave = threadIdx.x >> 5;
    const int lane = threadIdx.x & 31;
    const int bt   = blockIdx.y;
    const int tile = blockIdx.x * 4 + wave;
    const int i0   = tile << 4;
    if (i0 >= HW) return;                      // wave-uniform exit, EXEC stays full

    const int m  = lane & 15;                  // A-matrix row (position in tile)
    const int kb = (lane >> 4) << 3;           // A/B K base for this half-wave
    const int n  = lane & 15;                  // D-matrix column
    const int ic = min(i0 + m, HW - 1);        // clamped load position

    // per-lane base: element (c, ic) lives at base + c*HW (compile-time stride)
    const float* xb = x  + (size_t)bt * C * HW + ic;
    const float* pb = ps + (size_t)bt * C * HW + ic;

    v8f accq[4], accv[4];
#pragma unroll
    for (int nt = 0; nt < 4; ++nt) {
        accq[nt] = (v8f){0.f, 0.f, 0.f, 0.f, 0.f, 0.f, 0.f, 0.f};
        accv[nt] = (v8f){0.f, 0.f, 0.f, 0.f, 0.f, 0.f, 0.f, 0.f};
    }

    for (int kc = 0; kc < 12; ++kc) {
        const int c0 = (kc << 5) + kb;
        // ---- A tiles: gather 16 channels (streaming data: non-temporal) ----
        float xv[16], pv[16];
#pragma unroll
        for (int e = 0; e < 16; ++e) {
            const int c = c0 + e + (e < 8 ? 0 : 8);
            xv[e] = __builtin_nontemporal_load(xb + (size_t)c * HW);
            pv[e] = __builtin_nontemporal_load(pb + (size_t)c * HW);
        }
        v16bf ax, ap;
#pragma unroll
        for (int e = 0; e < 16; ++e) {
            ax[e] = (__bf16)xv[e];
            ap[e] = (__bf16)(xv[e] + pv[e]);
        }
        // ---- B tiles: load all 8 first, then a dense WMMA burst ----
        const unsigned short* bqk = bq + (kc << 11) + (lane << 4);
        const unsigned short* bvk = bv + (kc << 11) + (lane << 4);
        v16bf tbq[4], tbv[4];
#pragma unroll
        for (int nt = 0; nt < 4; ++nt) {
            tbq[nt] = *(const v16bf*)(bqk + (nt << 9));
            tbv[nt] = *(const v16bf*)(bvk + (nt << 9));
        }
#pragma unroll
        for (int nt = 0; nt < 4; ++nt) {
            accq[nt] = __builtin_amdgcn_wmma_f32_16x16x32_bf16(
                false, ap, false, tbq[nt], (short)0, accq[nt], false, false);
            accv[nt] = __builtin_amdgcn_wmma_f32_16x16x32_bf16(
                false, ax, false, tbv[nt], (short)0, accv[nt], false, false);
        }
    }

    // sim[i] = SCALE * dot(qk[i,:], cqk[:]) ; D layout: (M = r + 8*(lane>>4), N = lane&15)
    float sacc[8];
#pragma unroll
    for (int r = 0; r < 8; ++r) sacc[r] = 0.f;
#pragma unroll
    for (int nt = 0; nt < 4; ++nt) {
        const float cq = cqk[bt * DH + (nt << 4) + n];
#pragma unroll
        for (int r = 0; r < 8; ++r) sacc[r] += accq[nt][r] * cq;
    }
#pragma unroll
    for (int r = 0; r < 8; ++r) {
        float t = sacc[r];
        t += __shfl_xor(t, 1, 32);
        t += __shfl_xor(t, 2, 32);
        t += __shfl_xor(t, 4, 32);
        t += __shfl_xor(t, 8, 32);
        sacc[r] = t;
    }
    const int mb = (lane >> 4) << 3;
    if (n == 0) {
#pragma unroll
        for (int r = 0; r < 8; ++r) {
            int ii = i0 + mb + r;
            if (ii < HW) sim[bt * HW + ii] = SCALE * sacc[r];
        }
    }
    // store v tile: [bt][i][d]  (reread soon by pool kernel -> keep cacheable)
#pragma unroll
    for (int nt = 0; nt < 4; ++nt) {
#pragma unroll
        for (int r = 0; r < 8; ++r) {
            int ii = i0 + mb + r;
            if (ii < HW)
                vws[(((size_t)bt * HW + ii) << 6) + (nt << 4) + n] = accv[nt][r];
        }
    }
}

// ---------------------------------------------------------------
// Kernel C: per (b) softmax over HW, pool v, project -> aout.
// ---------------------------------------------------------------
__global__ __launch_bounds__(256)
void softmax_pool_kernel(const float* __restrict__ vws, float* __restrict__ simw,
                         const float* __restrict__ Wco, const float* __restrict__ bco,
                         float* __restrict__ aout, int hw, int s)
{
    const int b = blockIdx.x;
    const int t = threadIdx.x;
    __shared__ float red[256];
    __shared__ float pooled[DH];

    float* sb = simw + (size_t)b * hw;

    float mx = -1e30f;
    for (int i = t; i < hw; i += 256) mx = fmaxf(mx, sb[i]);
    red[t] = mx; __syncthreads();
    for (int o = 128; o > 0; o >>= 1) { if (t < o) red[t] = fmaxf(red[t], red[t + o]); __syncthreads(); }
    mx = red[0]; __syncthreads();

    float sum = 0.f;
    for (int i = t; i < hw; i += 256) { float e = __expf(sb[i] - mx); sb[i] = e; sum += e; }
    red[t] = sum; __syncthreads();
    for (int o = 128; o > 0; o >>= 1) { if (t < o) red[t] += red[t + o]; __syncthreads(); }
    const float inv = 1.f / red[0];
    __syncthreads();

    const int d = t & 63, g = t >> 6;
    float p = 0.f;
    for (int i = g; i < hw; i += 4)
        p += sb[i] * vws[(((size_t)b * hw + i) << 6) + d];
    red[t] = p * inv; __syncthreads();
    if (t < DH) pooled[t] = red[t] + red[t + 64] + red[t + 128] + red[t + 192];
    __syncthreads();

    for (int c = t; c < C; c += 256) {
        float o = bco[s * C + c];
        const float* w = Wco + (size_t)(s * C + c) * DH;
        for (int dd = 0; dd < DH; ++dd) o += pooled[dd] * w[dd];
        aout[b * C + c] = o;
    }
}

// ---------------------------------------------------------------
// Kernel V: vout[b,c,:,:] = outrow[b,c] broadcast.
// 130 MB of pure stores, never reread -> non-temporal 128-bit stores
// (native Clang ext_vector, which __builtin_nontemporal_store accepts).
// ---------------------------------------------------------------
__global__ __launch_bounds__(256)
void broadcast_vout_kernel(const float* __restrict__ outrow,
                           float* __restrict__ vout, int hw4)
{
    long id = (long)blockIdx.x * 256 + threadIdx.x;
    long total = (long)BT * C * hw4;
    if (id >= total) return;
    int row = (int)(id / hw4);                 // row = bt*C + c
    float val = outrow[row];
    v4f q = {val, val, val, val};
    __builtin_nontemporal_store(q, (v4f*)vout + id);
}

// ---------------------------------------------------------------
extern "C" void kernel_launch(void* const* d_in, const int* in_sizes, int n_in,
                              void* d_out, int out_size, void* d_ws, size_t ws_size,
                              hipStream_t stream)
{
    const float* fmap[4]; const float* posm[4]; const float* audio[4];
    for (int i = 0; i < 4; ++i) {
        fmap[i]  = (const float*)d_in[i * 3 + 0];
        posm[i]  = (const float*)d_in[i * 3 + 1];
        audio[i] = (const float*)d_in[i * 3 + 2];
    }
    const float* Wp   = (const float*)d_in[12];
    const float* bp   = (const float*)d_in[13];
    const float* pe   = (const float*)d_in[14];
    const float* Wqk  = (const float*)d_in[15];
    const float* Wcqk = (const float*)d_in[16];
    const float* Wv   = (const float*)d_in[17];
    const float* Wcv  = (const float*)d_in[18];
    const float* Wo   = (const float*)d_in[19];
    const float* bo   = (const float*)d_in[20];
    const float* Wco  = (const float*)d_in[21];
    const float* bco  = (const float*)d_in[22];
    float* out = (float*)d_out;

    // workspace layout (bytes): total ~17.1 MB
    char* ws = (char*)d_ws;
    unsigned short* brep   = (unsigned short*)ws;            // 196608 bf16 = 393216 B
    float* cqk_ws    = (float*)(ws + 393216);                // NS*BT*DH f32 = 10240 B
    float* outrow_ws = (float*)(ws + 403456);                // NS*BT*C  f32 = 61440 B
    float* sim_ws    = (float*)(ws + 464896);                // BT*6400  f32 = 256000 B (reused per stage)
    float* v_ws      = (float*)(ws + 720896);                // BT*6400*64 f32 = 16384000 B (reused)

    static const int  HW[4]   = {6400, 1600, 400, 100};
    static const long VOFF[4] = {0L, 24576000L, 30720000L, 32256000L};
    const long AOFF = 32640000L;

    repack_kernel<<<dim3(768), dim3(256), 0, stream>>>(Wqk, Wv, brep);
    prep_kernel<<<dim3(NS, BT), dim3(384), 0, stream>>>(
        audio[0], audio[1], audio[2], audio[3],
        Wp, bp, pe, Wcqk, Wcv, Wo, bo, cqk_ws, outrow_ws);

    for (int s = 0; s < NS; ++s) {
        const int hw = HW[s];
        const int tiles = (hw + 15) / 16;
        dim3 gB((tiles + 3) / 4, BT);
        const unsigned short* bqs = brep + (size_t)(s * 2 + 0) * 24576;
        const unsigned short* bvs = brep + (size_t)(s * 2 + 1) * 24576;
        const float* cq = cqk_ws + s * BT * DH;

        switch (s) {
        case 0: stage_qkv_kernel<6400><<<gB, dim3(128), 0, stream>>>(
                    fmap[s], posm[s], bqs, bvs, cq, sim_ws, v_ws); break;
        case 1: stage_qkv_kernel<1600><<<gB, dim3(128), 0, stream>>>(
                    fmap[s], posm[s], bqs, bvs, cq, sim_ws, v_ws); break;
        case 2: stage_qkv_kernel<400><<<gB, dim3(128), 0, stream>>>(
                    fmap[s], posm[s], bqs, bvs, cq, sim_ws, v_ws); break;
        default: stage_qkv_kernel<100><<<gB, dim3(128), 0, stream>>>(
                    fmap[s], posm[s], bqs, bvs, cq, sim_ws, v_ws); break;
        }

        softmax_pool_kernel<<<dim3(BT), dim3(256), 0, stream>>>(
            v_ws, sim_ws, Wco, bco, out + AOFF + (long)s * BT * C, hw, s);

        const long total4 = (long)BT * C * (hw / 4);
        broadcast_vout_kernel<<<dim3((unsigned)((total4 + 255) / 256)), dim3(256), 0, stream>>>(
            outrow_ws + (size_t)s * BT * C, out + VOFF[s], hw / 4);
    }
}
